// BambaMambaDecoderLayer_22797686407730
// MI455X (gfx1250) — matrix operand order repaired and checked by reading, not compiled, not run on hardware
//
#include <hip/hip_runtime.h>
#include <hip/hip_bf16.h>
#include <cstdint>
#include <cstddef>

// ---- problem dims ----
#define D_MODEL 2048
#define D_INNER 4096
#define N_HEADS 64
#define D_HEAD  64
#define D_STATE 128
#define K_CONV  4
#define D_FF    8192
#define CONV_DIM (D_INNER + 2 * D_STATE)             // 4352
#define D_PROJ   (2 * D_INNER + 2 * D_STATE + N_HEADS) // 8512
#define BATCH 2
#define SEQ   2048
#define NTOK  (BATCH * SEQ)                           // 4096
#define EPS_F 1e-5f

typedef __attribute__((ext_vector_type(16))) __bf16 v16bf;
typedef __attribute__((ext_vector_type(8)))  float  v8f;

// =====================================================================
// WMMA bf16 GEMM: C[M,N] (f32) = A[M,K] (bf16, row-major) * Bt[N,K]^T
// Bt is B stored transposed (N x K, row-major) so both operands stage
// into LDS with contiguous K.
// Block: 256 threads (8 waves, 2x4). Tile: 128(M) x 256(N), K-step 32.
// Each wave: 64x64 sub-tile = 4x4 grid of 16x16x32 WMMA ops
// (16 ds_load_b128 : 16 v_wmma per K-step -> 1.0 LDS reads/WMMA).
// =====================================================================
__global__ __launch_bounds__(256) void gemm_bf16_wmma(
    const __bf16* __restrict__ A,
    const __bf16* __restrict__ Bt,
    float* __restrict__ C,
    int M, int N, int K)
{
    constexpr int TK = 32;
    constexpr int LSTR = 40; // padded row stride (bf16 units) to spread banks
    __shared__ __align__(16) __bf16 As[128 * LSTR];
    __shared__ __align__(16) __bf16 Bs[256 * LSTR];

    const int tid = threadIdx.x;
    const int m0  = blockIdx.y * 128;
    const int n0  = blockIdx.x * 256;

    // staging: 2 threads per tile row, each covers 16 bf16 (2x b128).
    // A: rows 0..127.  B: rows (tid>>1) and (tid>>1)+128.
    const int ldRow = tid >> 1;
    const int ldK   = (tid & 1) * 16;

    const int wave = tid >> 5;
    const int lane = tid & 31;
    const int hl   = lane >> 4;   // half-wave select
    const int l16  = lane & 15;
    const int wm   = (wave & 1) * 64;   // 0 / 64
    const int wn   = (wave >> 1) * 64;  // 0 / 64 / 128 / 192

    const v8f vzero = {0.f,0.f,0.f,0.f,0.f,0.f,0.f,0.f};
    v8f acc[4][4];
#pragma unroll
    for (int i = 0; i < 4; ++i)
#pragma unroll
        for (int j = 0; j < 4; ++j) acc[i][j] = vzero;

    const size_t aRowOff = (size_t)(m0 + ldRow) * (size_t)K;
    const int    gn0     = n0 + ldRow;
    const int    gn1     = n0 + ldRow + 128;
    const bool   bValid0 = (gn0 < N);
    const bool   bValid1 = (gn1 < N);
    const size_t bRow0   = (size_t)(bValid0 ? gn0 : 0) * (size_t)K;
    const size_t bRow1   = (size_t)(bValid1 ? gn1 : 0) * (size_t)K;

    union FragU { v16bf v; uint4 q[2]; };

    for (int k0 = 0; k0 < K; k0 += TK) {
        // ---- stage A tile ----
        const uint4* ap = (const uint4*)(A + aRowOff + k0 + ldK);
        uint4 a0 = ap[0];
        uint4 a1 = ap[1];
        *(uint4*)(&As[ldRow * LSTR + ldK    ]) = a0;
        *(uint4*)(&As[ldRow * LSTR + ldK + 8]) = a1;

        // ---- stage B tile (two rows/thread; zero-fill out-of-range N) ----
        uint4 b0 = make_uint4(0u,0u,0u,0u), b1 = make_uint4(0u,0u,0u,0u);
        uint4 c0 = make_uint4(0u,0u,0u,0u), c1 = make_uint4(0u,0u,0u,0u);
        if (bValid0) {
            const uint4* bp = (const uint4*)(Bt + bRow0 + k0 + ldK);
            b0 = bp[0]; b1 = bp[1];
        }
        if (bValid1) {
            const uint4* bp = (const uint4*)(Bt + bRow1 + k0 + ldK);
            c0 = bp[0]; c1 = bp[1];
        }
        *(uint4*)(&Bs[ldRow * LSTR + ldK            ]) = b0;
        *(uint4*)(&Bs[ldRow * LSTR + ldK + 8        ]) = b1;
        *(uint4*)(&Bs[(ldRow + 128) * LSTR + ldK    ]) = c0;
        *(uint4*)(&Bs[(ldRow + 128) * LSTR + ldK + 8]) = c1;

        if (k0 + TK < K) { // hint next tiles into cache (global_prefetch_b8)
            __builtin_prefetch(A + aRowOff + k0 + TK + ldK, 0, 0);
            __builtin_prefetch(Bt + bRow0 + k0 + TK + ldK, 0, 0);
            __builtin_prefetch(Bt + bRow1 + k0 + TK + ldK, 0, 0);
        }
        __syncthreads();

        // ---- build fragments ----
        // A 16-bit 16x32 layout: half0 lanes hold K {0..7,16..23},
        //                        half1 lanes hold K {8..15,24..31}
        FragU fa[4];
#pragma unroll
        for (int mt = 0; mt < 4; ++mt) {
            const int r = wm + mt * 16 + l16;
            fa[mt].q[0] = *(const uint4*)(&As[r * LSTR + hl * 8]);
            fa[mt].q[1] = *(const uint4*)(&As[r * LSTR + 16 + hl * 8]);
        }
        // B 16-bit 32x16 layout: half0 lanes hold K {0..7,16..23}, half1 K {8..15,24..31}
        FragU fb[4];
#pragma unroll
        for (int nt = 0; nt < 4; ++nt) {
            const int r = wn + nt * 16 + l16;
            fb[nt].q[0] = *(const uint4*)(&Bs[r * LSTR + hl * 8]);
            fb[nt].q[1] = *(const uint4*)(&Bs[r * LSTR + 16 + hl * 8]);
        }

#pragma unroll
        for (int mt = 0; mt < 4; ++mt)
#pragma unroll
            for (int nt = 0; nt < 4; ++nt)
                acc[mt][nt] = __builtin_amdgcn_wmma_f32_16x16x32_bf16(
                    false, fa[mt].v, false, fb[nt].v,
                    (short)0, acc[mt][nt], false, false);
        __syncthreads();
    }

    // ---- writeback: VGPR r holds M=r (lanes 0-15) / M=r+8 (lanes 16-31) ----
#pragma unroll
    for (int mt = 0; mt < 4; ++mt) {
#pragma unroll
        for (int nt = 0; nt < 4; ++nt) {
            const int n = n0 + wn + nt * 16 + l16;
            if (n < N) {
#pragma unroll
                for (int r = 0; r < 8; ++r) {
                    const int m = m0 + wm + mt * 16 + r + hl * 8;
                    C[(size_t)m * (size_t)N + n] = acc[mt][nt][r];
                }
            }
        }
    }
}

// ---- fp32 weight [K,N] -> bf16 transposed [N,K] ----
__global__ void convert_transpose_kernel(const float* __restrict__ W,
                                         __bf16* __restrict__ Wt, int K, int N)
{
    size_t idx = (size_t)blockIdx.x * 256 + threadIdx.x;
    if (idx >= (size_t)K * N) return;
    int n = (int)(idx / K), k = (int)(idx % K);
    Wt[idx] = (__bf16)W[(size_t)k * N + n];
}

// ---- RMSNorm, bf16 output ----
__global__ void rmsnorm_bf16_kernel(const float* __restrict__ x,
                                    const float* __restrict__ w,
                                    __bf16* __restrict__ out, int D)
{
    __shared__ float red[256];
    const int t = blockIdx.x;
    const float* xp = x + (size_t)t * D;
    float s = 0.f;
    for (int i = threadIdx.x; i < D; i += 256) { float v = xp[i]; s += v * v; }
    red[threadIdx.x] = s; __syncthreads();
    for (int off = 128; off > 0; off >>= 1) {
        if (threadIdx.x < off) red[threadIdx.x] += red[threadIdx.x + off];
        __syncthreads();
    }
    const float rs = rsqrtf(red[0] / D + EPS_F);
    for (int i = threadIdx.x; i < D; i += 256)
        out[(size_t)t * D + i] = (__bf16)(xp[i] * rs * w[i]);
}

// ---- causal depthwise conv (K=4) + bias + silu ----
__global__ void conv_silu_kernel(const float* __restrict__ zxbcdt,
                                 const float* __restrict__ cw,
                                 const float* __restrict__ cb,
                                 float* __restrict__ xconv)
{
    size_t idx = (size_t)blockIdx.x * 256 + threadIdx.x;
    if (idx >= (size_t)NTOK * CONV_DIM) return;
    const int c  = (int)(idx % CONV_DIM);
    const int bt = (int)(idx / CONV_DIM);
    const int t  = bt % SEQ;
    const float* base = zxbcdt + (size_t)bt * D_PROJ + D_INNER + c;
    float acc = cb[c];
#pragma unroll
    for (int k = 0; k < K_CONV; ++k) {
        const int dt = t - (K_CONV - 1) + k;
        if (dt >= 0) acc += cw[c * K_CONV + k] * base[(ptrdiff_t)(k - (K_CONV - 1)) * D_PROJ];
    }
    xconv[idx] = acc / (1.f + expf(-acc));
}

// ---- dt = softplus(dt_raw + dt_bias) ----
__global__ void dt_softplus_kernel(const float* __restrict__ zxbcdt,
                                   const float* __restrict__ dt_bias,
                                   float* __restrict__ dtp)
{
    int idx = blockIdx.x * 256 + threadIdx.x;
    if (idx >= NTOK * N_HEADS) return;
    const int h  = idx % N_HEADS;
    const int bt = idx / N_HEADS;
    const float x = zxbcdt[(size_t)bt * D_PROJ + (D_PROJ - N_HEADS) + h] + dt_bias[h];
    dtp[idx] = (x > 20.f) ? x : log1pf(expf(x));
}

// ---- selective scan: one block per (batch, head); state in registers ----
__global__ __launch_bounds__(256) void ssm_scan_kernel(
    const float* __restrict__ xconv, const float* __restrict__ dtp,
    const float* __restrict__ A_log, const float* __restrict__ Dv,
    float* __restrict__ y)
{
    const int b   = blockIdx.x >> 6;
    const int h   = blockIdx.x & 63;
    const int tid = threadIdx.x;
    const int p   = tid >> 2;            // 0..63 head dim
    const int nc  = (tid & 3) * 32;      // 32-wide state slice
    const float Ah = -expf(A_log[h]);
    const float Dh = Dv[h];

    __shared__ float sB[D_STATE], sC[D_STATE], sX[D_HEAD];

    float s[32];
#pragma unroll
    for (int i = 0; i < 32; ++i) s[i] = 0.f;

    for (int t = 0; t < SEQ; ++t) {
        const size_t bt = (size_t)b * SEQ + t;
        const float* row = xconv + bt * CONV_DIM;
        __syncthreads();
        if (tid < D_STATE) sB[tid] = row[D_INNER + tid];
        else               sC[tid - D_STATE] = row[D_INNER + tid];
        if (tid < D_HEAD)  sX[tid] = row[(size_t)h * D_HEAD + tid];
        const float dtv = dtp[bt * N_HEADS + h];
        const float dA  = __expf(dtv * Ah);
        __syncthreads();

        const float xp   = sX[p];
        const float coef = dtv * xp;
        float part = 0.f;
#pragma unroll
        for (int i = 0; i < 32; ++i) {
            s[i] = s[i] * dA + coef * sB[nc + i];
            part += s[i] * sC[nc + i];
        }
        part += __shfl_xor(part, 1);
        part += __shfl_xor(part, 2);
        if ((tid & 3) == 0)
            y[bt * D_INNER + (size_t)h * D_HEAD + p] = part + Dh * xp;
    }
}

// ---- y_gated = rms(y * silu(z)) * gnorm_w, bf16 out ----
__global__ void gate_rms_kernel(const float* __restrict__ y,
                                const float* __restrict__ zxbcdt,
                                const float* __restrict__ gw,
                                __bf16* __restrict__ out)
{
    __shared__ float red[256];
    const int t = blockIdx.x;
    const float* yp = y + (size_t)t * D_INNER;
    const float* zp = zxbcdt + (size_t)t * D_PROJ;
    float s = 0.f;
    for (int i = threadIdx.x; i < D_INNER; i += 256) {
        const float z = zp[i];
        const float v = yp[i] * (z / (1.f + expf(-z)));
        s += v * v;
    }
    red[threadIdx.x] = s; __syncthreads();
    for (int off = 128; off > 0; off >>= 1) {
        if (threadIdx.x < off) red[threadIdx.x] += red[threadIdx.x + off];
        __syncthreads();
    }
    const float rs = rsqrtf(red[0] / D_INNER + EPS_F);
    for (int i = threadIdx.x; i < D_INNER; i += 256) {
        const float z = zp[i];
        const float v = yp[i] * (z / (1.f + expf(-z)));
        out[(size_t)t * D_INNER + i] = (__bf16)(v * rs * gw[i]);
    }
}

// ---- ff = silu(gate) * up, bf16 out ----
__global__ void silu_mul_bf16_kernel(const float* __restrict__ g,
                                     const float* __restrict__ u,
                                     __bf16* __restrict__ out, size_t n)
{
    size_t i = (size_t)blockIdx.x * 256 + threadIdx.x;
    if (i >= n) return;
    const float x = g[i];
    out[i] = (__bf16)((x / (1.f + expf(-x))) * u[i]);
}

// ---- o = a + b ----
__global__ void add_kernel(const float* __restrict__ a, const float* __restrict__ b,
                           float* __restrict__ o, size_t n)
{
    size_t i = (size_t)blockIdx.x * 256 + threadIdx.x;
    if (i < n) o[i] = a[i] + b[i];
}

static inline unsigned ceil_div(size_t a, size_t b) { return (unsigned)((a + b - 1) / b); }

extern "C" void kernel_launch(void* const* d_in, const int* in_sizes, int n_in,
                              void* d_out, int out_size, void* d_ws, size_t ws_size,
                              hipStream_t stream)
{
    (void)in_sizes; (void)n_in; (void)out_size; (void)ws_size;
    const float* hidden    = (const float*)d_in[0];
    const float* ln1_w     = (const float*)d_in[1];
    const float* ln2_w     = (const float*)d_in[2];
    const float* in_proj_w = (const float*)d_in[3];
    const float* conv_w    = (const float*)d_in[4];
    const float* conv_b    = (const float*)d_in[5];
    const float* dt_bias   = (const float*)d_in[6];
    const float* A_log     = (const float*)d_in[7];
    const float* Dv        = (const float*)d_in[8];
    const float* gnorm_w   = (const float*)d_in[9];
    const float* out_proj_w= (const float*)d_in[10];
    const float* gate_w    = (const float*)d_in[11];
    const float* up_w      = (const float*)d_in[12];
    const float* down_w    = (const float*)d_in[13];
    float* out = (float*)d_out;

    char* ws = (char*)d_ws;
    size_t cur = 0;
    auto alloc = [&](size_t bytes) { size_t o = cur; cur += (bytes + 255) & ~(size_t)255; return o; };

    const size_t o_wT_in   = alloc((size_t)D_PROJ * D_MODEL * 2);
    const size_t o_wT_out  = alloc((size_t)D_MODEL * D_INNER * 2);
    const size_t o_wT_gate = alloc((size_t)D_FF * D_MODEL * 2);
    const size_t o_wT_up   = alloc((size_t)D_FF * D_MODEL * 2);
    const size_t o_wT_down = alloc((size_t)D_MODEL * D_FF * 2);
    const size_t o_hnorm   = alloc((size_t)NTOK * D_MODEL * 2);
    const size_t o_zxbcdt  = alloc((size_t)NTOK * D_PROJ * 4);
    const size_t o_xconv   = alloc((size_t)NTOK * CONV_DIM * 4);
    const size_t o_y       = alloc((size_t)NTOK * D_INNER * 4);
    const size_t o_dtp     = alloc((size_t)NTOK * N_HEADS * 4);
    const size_t o_ygate   = alloc((size_t)NTOK * D_INNER * 2);
    const size_t o_mamba   = alloc((size_t)NTOK * D_MODEL * 4);
    const size_t o_h       = alloc((size_t)NTOK * D_MODEL * 4);
    const size_t o_h2n     = alloc((size_t)NTOK * D_MODEL * 2);
    // aliased regions (lifetimes do not overlap):
    const size_t o_gateout = o_zxbcdt;            // 134MB <= 139MB
    const size_t o_upout   = o_xconv;             // 134MB <= 138MB (xconv+y)
    const size_t o_ff      = o_ygate;             // 67MB == ygate+mamba
    const size_t o_downout = o_wT_in;             // 33.5MB <= 34.9MB

    __bf16* wT_in    = (__bf16*)(ws + o_wT_in);
    __bf16* wT_out   = (__bf16*)(ws + o_wT_out);
    __bf16* wT_gate  = (__bf16*)(ws + o_wT_gate);
    __bf16* wT_up    = (__bf16*)(ws + o_wT_up);
    __bf16* wT_down  = (__bf16*)(ws + o_wT_down);
    __bf16* hnorm    = (__bf16*)(ws + o_hnorm);
    float*  zxbcdt   = (float*)(ws + o_zxbcdt);
    float*  xconv    = (float*)(ws + o_xconv);
    float*  ybuf     = (float*)(ws + o_y);
    float*  dtp      = (float*)(ws + o_dtp);
    __bf16* ygate    = (__bf16*)(ws + o_ygate);
    float*  mamba    = (float*)(ws + o_mamba);
    float*  hres     = (float*)(ws + o_h);
    __bf16* h2n      = (__bf16*)(ws + o_h2n);
    float*  gate_out = (float*)(ws + o_gateout);
    float*  up_out   = (float*)(ws + o_upout);
    __bf16* ff       = (__bf16*)(ws + o_ff);
    float*  down_out = (float*)(ws + o_downout);

    // ---- weight conversions (fp32 [K,N] -> bf16 [N,K]) ----
    convert_transpose_kernel<<<ceil_div((size_t)D_MODEL * D_PROJ, 256), 256, 0, stream>>>(in_proj_w, wT_in, D_MODEL, D_PROJ);
    convert_transpose_kernel<<<ceil_div((size_t)D_INNER * D_MODEL, 256), 256, 0, stream>>>(out_proj_w, wT_out, D_INNER, D_MODEL);
    convert_transpose_kernel<<<ceil_div((size_t)D_MODEL * D_FF, 256), 256, 0, stream>>>(gate_w, wT_gate, D_MODEL, D_FF);
    convert_transpose_kernel<<<ceil_div((size_t)D_MODEL * D_FF, 256), 256, 0, stream>>>(up_w, wT_up, D_MODEL, D_FF);
    convert_transpose_kernel<<<ceil_div((size_t)D_FF * D_MODEL, 256), 256, 0, stream>>>(down_w, wT_down, D_FF, D_MODEL);

    // ---- Mamba2 branch ----
    rmsnorm_bf16_kernel<<<NTOK, 256, 0, stream>>>(hidden, ln1_w, hnorm, D_MODEL);
    gemm_bf16_wmma<<<dim3(ceil_div(D_PROJ, 256), NTOK / 128), 256, 0, stream>>>(hnorm, wT_in, zxbcdt, NTOK, D_PROJ, D_MODEL);
    conv_silu_kernel<<<ceil_div((size_t)NTOK * CONV_DIM, 256), 256, 0, stream>>>(zxbcdt, conv_w, conv_b, xconv);
    dt_softplus_kernel<<<ceil_div((size_t)NTOK * N_HEADS, 256), 256, 0, stream>>>(zxbcdt, dt_bias, dtp);
    ssm_scan_kernel<<<BATCH * N_HEADS, 256, 0, stream>>>(xconv, dtp, A_log, Dv, ybuf);
    gate_rms_kernel<<<NTOK, 256, 0, stream>>>(ybuf, zxbcdt, gnorm_w, ygate);
    gemm_bf16_wmma<<<dim3(ceil_div(D_MODEL, 256), NTOK / 128), 256, 0, stream>>>(ygate, wT_out, mamba, NTOK, D_MODEL, D_INNER);
    add_kernel<<<ceil_div((size_t)NTOK * D_MODEL, 256), 256, 0, stream>>>(hidden, mamba, hres, (size_t)NTOK * D_MODEL);

    // ---- MLP branch ----
    rmsnorm_bf16_kernel<<<NTOK, 256, 0, stream>>>(hres, ln2_w, h2n, D_MODEL);
    gemm_bf16_wmma<<<dim3(ceil_div(D_FF, 256), NTOK / 128), 256, 0, stream>>>(h2n, wT_gate, gate_out, NTOK, D_FF, D_MODEL);
    gemm_bf16_wmma<<<dim3(ceil_div(D_FF, 256), NTOK / 128), 256, 0, stream>>>(h2n, wT_up, up_out, NTOK, D_FF, D_MODEL);
    silu_mul_bf16_kernel<<<ceil_div((size_t)NTOK * D_FF, 256), 256, 0, stream>>>(gate_out, up_out, ff, (size_t)NTOK * D_FF);
    gemm_bf16_wmma<<<dim3(ceil_div(D_MODEL, 256), NTOK / 128), 256, 0, stream>>>(ff, wT_down, down_out, NTOK, D_MODEL, D_FF);
    add_kernel<<<ceil_div((size_t)NTOK * D_MODEL, 256), 256, 0, stream>>>(hres, down_out, out, (size_t)NTOK * D_MODEL);
}
